// LinearAttention_59734405153127
// MI455X (gfx1250) — matrix operand order repaired and checked by reading, not compiled, not run on hardware
//
#include <hip/hip_runtime.h>
#include <hip/hip_bf16.h>

// ---------------------------------------------------------------------------
// Linear attention (B=4, S=4096, E=D=1024) for gfx1250 / MI455X.
// GEMMs on v_wmma_f32_16x16x32_bf16 (fp32 accumulate); bf16 intermediates to
// halve L2/HBM traffic (workload is bandwidth-bound at 23.3 TB/s).  Shared
// K x N B-panels are staged into LDS with the Tensor Data Mover
// (tensor_load_to_lds + s_wait_tensorcnt) so one async DMA per 128-deep K
// chunk replaces 8 waves' worth of strided column gathers.
// Workspace layout (~126 MB):
//   [0,32MB)    q   bf16 (B*S, D)
//   [32,64MB)   k^T bf16 (B, D, S)
//   [64,96MB)   v   bf16 (B*S, D)
//   [96,112MB)  num f32 (B, D, D)
//   [112,120MB) ctx bf16 (B, D, D)
//   [120MB,+16KB) den f32 (B, D)
// ---------------------------------------------------------------------------

typedef __attribute__((ext_vector_type(16))) __bf16 v16bf;
typedef __attribute__((ext_vector_type(8)))  __bf16 v8bf;
typedef __attribute__((ext_vector_type(8)))  float  v8f;
typedef __attribute__((ext_vector_type(4))) unsigned int u32x4;
typedef __attribute__((ext_vector_type(4))) int          i32x4;
typedef __attribute__((ext_vector_type(8))) int          i32x8;

static constexpr int BB = 4;
static constexpr int SS = 4096;
static constexpr int EE = 1024;
static constexpr int DD = 1024;
static constexpr int KCH = 128;   // K-chunk staged per TDM transfer

#ifndef __has_builtin
#define __has_builtin(x) 0
#endif
#if defined(__AMDGCN__) && __has_builtin(__builtin_amdgcn_tensor_load_to_lds)
#define LA_HAVE_TDM 1
#else
#define LA_HAVE_TDM 0
#endif
#if defined(__has_include)
#if __has_include(<hip/amd_detail/amd_gfx1250_TDM.h>)
#define LA_TDM_6ARG 1
#endif
#endif
#ifndef LA_TDM_6ARG
#define LA_TDM_6ARG 0
#endif

// ---- WMMA helper ----------------------------------------------------------
__device__ __forceinline__ v8f wmma_bf16(v16bf a, v16bf b, v8f c) {
  // (neg_a, A, neg_b, B, c_mod, C, reuse_a, reuse_b)
  return __builtin_amdgcn_wmma_f32_16x16x32_bf16(false, a, false, b,
                                                 (short)0, c, false, false);
}

// ---- Tensor Data Mover: 2D tile Global -> LDS (D# per ISA 08 §8.3/8.4) ----
#if LA_HAVE_TDM
__device__ __forceinline__ void tdm_load_2d(
    unsigned lds_byte_off, const void* gaddr,
    unsigned data_size_code,   // 0=1B 1=2B 2=4B
    unsigned tile_cols,        // tile_dim0 (elements)
    unsigned tile_rows,        // tile_dim1
    unsigned row_stride_elems) // tensor_dim0_stride
{
  unsigned long long ga = (unsigned long long)(uintptr_t)gaddr;
  u32x4 g0;
  g0[0] = 1u;                                        // count=1, user descriptor
  g0[1] = lds_byte_off;                              // lds_addr
  g0[2] = (unsigned)(ga & 0xffffffffu);              // global_addr[31:0]
  g0[3] = (unsigned)((ga >> 32) & 0x01ffffffu) | (2u << 30);  // addr[56:32]|type=2
  i32x8 g1;
  g1[0] = (int)(data_size_code << 16);               // wg_mask=0, data_size
  g1[1] = (int)((tile_cols & 0xffffu) << 16);        // tensor_dim0 = tile_cols
  g1[2] = (int)((tile_rows & 0xffffu) << 16);        // tensor_dim1 = tile_rows
  g1[3] = (int)(tile_cols << 16);                    // tile_dim0 (dim1 hi = 0)
  g1[4] = (int)(tile_rows & 0xffffu);                // tile_dim1 (tile_dim2=0)
  g1[5] = (int)row_stride_elems;                     // tensor_dim0_stride lo
  g1[6] = 0;                                         // stride hi / dim1_stride
  g1[7] = 0;
  i32x4 z4 = {0, 0, 0, 0};
#if LA_TDM_6ARG
  i32x8 z8 = {0, 0, 0, 0, 0, 0, 0, 0};
  __builtin_amdgcn_tensor_load_to_lds(g0, g1, z4, z4, z8, 0);
#else
  __builtin_amdgcn_tensor_load_to_lds(g0, g1, z4, z4, 0);
#endif
  __builtin_amdgcn_s_wait_tensorcnt(0);
}
#endif

// A fragment, 16x32 bf16, source row-major (ld = row stride in elements).
__device__ __forceinline__ v16bf load_a_bf16(const __bf16* __restrict__ base,
                                             int ld, int m, int k0, int half) {
  const __bf16* p = base + (size_t)m * ld + k0 + half * 16;
  v8bf lo = *(const v8bf*)(p);
  v8bf hi = *(const v8bf*)(p + 8);
  v16bf a;
#pragma unroll
  for (int i = 0; i < 8; ++i) { a[i] = lo[i]; a[8 + i] = hi[i]; }
  return a;
}

// B fragment from an LDS-resident K x 32 bf16 tile: b[idx] = tile[(kk+kb+idx)*32+col]
__device__ __forceinline__ v16bf lds_b_frag_bf16(const __bf16* tile, int kk,
                                                 int col, int half) {
  const __bf16* p = tile + (size_t)(kk + half * 16) * 32 + col;
  v16bf b;
#pragma unroll
  for (int i = 0; i < 16; ++i) b[i] = p[(size_t)i * 32];
  return b;
}

// B fragment from an LDS-resident K x 32 fp32 tile (converted to bf16).
__device__ __forceinline__ v16bf lds_b_frag_f32(const float* tile, int kk,
                                                int col, int half) {
  const float* p = tile + (size_t)(kk + half * 16) * 32 + col;
  v16bf b;
#pragma unroll
  for (int i = 0; i < 16; ++i) b[i] = (__bf16)p[(size_t)i * 32];
  return b;
}

// Fallback direct-global B gathers (no TDM available).
__device__ __forceinline__ v16bf load_b_gather_bf16(const __bf16* __restrict__ base,
                                                    int ld, int n, int k0, int half) {
  const __bf16* p = base + (size_t)(k0 + half * 16) * ld + n;
  v16bf b;
#pragma unroll
  for (int i = 0; i < 16; ++i) b[i] = p[(size_t)i * ld];
  return b;
}
__device__ __forceinline__ v16bf load_b_gather_f32(const float* __restrict__ base,
                                                   int ld, int n, int k0, int half) {
  const float* p = base + (size_t)(k0 + half * 16) * ld + n;
  v16bf b;
#pragma unroll
  for (int i = 0; i < 16; ++i) b[i] = (__bf16)p[(size_t)i * ld];
  return b;
}

// B fragment where B[k][n] = src[n*ld + k]: per-lane contiguous 32B load.
__device__ __forceinline__ v16bf load_b_contig_bf16(const __bf16* __restrict__ base,
                                                    size_t rowoff, int k0, int half) {
  const __bf16* p = base + rowoff + k0 + half * 16;
  v8bf lo = *(const v8bf*)(p);
  v8bf hi = *(const v8bf*)(p + 8);
  v16bf b;
#pragma unroll
  for (int i = 0; i < 8; ++i) { b[i] = lo[i]; b[8 + i] = hi[i]; }
  return b;
}

// ---------------------------------------------------------------------------
// Kernel 1: q/k/v projections.  grid = (D/32, (B*S)/128, 3)
// ---------------------------------------------------------------------------
__global__ __launch_bounds__(256) void la_qkv_kernel(
    const float* __restrict__ X, const float* __restrict__ Wq,
    const float* __restrict__ Wk, const float* __restrict__ Wv,
    __bf16* __restrict__ qO, __bf16* __restrict__ kT, __bf16* __restrict__ vO) {
  const int mode = blockIdx.z;
  const float* W = (mode == 0) ? Wq : ((mode == 1) ? Wk : Wv);
  const int wave = threadIdx.x >> 5;
  const int lane = threadIdx.x & 31;
  const int half = lane >> 4;
  const int l16  = lane & 15;
  const int m0 = (blockIdx.y * 8 + wave) * 16;
  const int n0 = blockIdx.x * 32;

#if LA_HAVE_TDM
  __shared__ float tileW[KCH * 32];   // 16 KB fp32 W panel shared by 8 waves
#endif

  v8f c0 = {}; v8f c1 = {};
  for (int k0 = 0; k0 < EE; k0 += KCH) {
#if LA_HAVE_TDM
    __syncthreads();                       // previous panel fully consumed
    if (wave == 0) {                       // one async DMA for the whole block
      tdm_load_2d((unsigned)(uintptr_t)tileW, W + (size_t)k0 * DD + n0,
                  /*4B*/2, /*cols*/32, /*rows*/KCH, /*stride*/DD);
    }
    __syncthreads();                       // panel visible to all waves
#endif
#pragma unroll
    for (int kk = 0; kk < KCH; kk += 32) {
      const int kg = k0 + kk;
      // A fragment from fp32 X, converted to bf16.
      const float* pa = X + (size_t)(m0 + l16) * EE + kg + half * 16;
      float4 f0 = *(const float4*)(pa);
      float4 f1 = *(const float4*)(pa + 4);
      float4 f2 = *(const float4*)(pa + 8);
      float4 f3 = *(const float4*)(pa + 12);
      v16bf a;
      a[0]  = (__bf16)f0.x; a[1]  = (__bf16)f0.y; a[2]  = (__bf16)f0.z; a[3]  = (__bf16)f0.w;
      a[4]  = (__bf16)f1.x; a[5]  = (__bf16)f1.y; a[6]  = (__bf16)f1.z; a[7]  = (__bf16)f1.w;
      a[8]  = (__bf16)f2.x; a[9]  = (__bf16)f2.y; a[10] = (__bf16)f2.z; a[11] = (__bf16)f2.w;
      a[12] = (__bf16)f3.x; a[13] = (__bf16)f3.y; a[14] = (__bf16)f3.z; a[15] = (__bf16)f3.w;
#if LA_HAVE_TDM
      v16bf b0 = lds_b_frag_f32(tileW, kk, l16, half);
      v16bf b1 = lds_b_frag_f32(tileW, kk, l16 + 16, half);
#else
      v16bf b0 = load_b_gather_f32(W, DD, n0 + l16, kg, half);
      v16bf b1 = load_b_gather_f32(W, DD, n0 + 16 + l16, kg, half);
#endif
      c0 = wmma_bf16(a, b0, c0);
      c1 = wmma_bf16(a, b1, c1);
    }
  }

#pragma unroll
  for (int r = 0; r < 8; ++r) {
    const int m = m0 + r + half * 8;       // global row in (B*S)
    float x0 = c0[r], x1 = c1[r];
    if (mode < 2) {                        // relu + 1 for q and k
      x0 = fmaxf(x0, 0.0f) + 1.0f;
      x1 = fmaxf(x1, 0.0f) + 1.0f;
    }
    const int nA = n0 + l16, nB = n0 + 16 + l16;
    if (mode == 1) {                       // k stored transposed (B, D, S)
      const int b = m >> 12;               // S = 4096
      const int s = m & (SS - 1);
      kT[((size_t)b * DD + nA) * SS + s] = (__bf16)x0;
      kT[((size_t)b * DD + nB) * SS + s] = (__bf16)x1;
    } else {
      __bf16* O = (mode == 0) ? qO : vO;
      O[(size_t)m * DD + nA] = (__bf16)x0;
      O[(size_t)m * DD + nB] = (__bf16)x1;
    }
  }
}

// ---------------------------------------------------------------------------
// Kernel 2: den[b,x] = sum_s k[b,s,x]  (contiguous rows of kT). grid = B*D.
// ---------------------------------------------------------------------------
__global__ __launch_bounds__(256) void la_den_kernel(
    const __bf16* __restrict__ kT, float* __restrict__ den) {
  const int row = blockIdx.x;                        // b*D + x
  const __bf16* p = kT + (size_t)row * SS;
  float s = 0.0f;
  for (int i = threadIdx.x; i < SS; i += 256) s += (float)p[i];
  __shared__ float red[256];
  red[threadIdx.x] = s;
  __syncthreads();
  for (int off = 128; off > 0; off >>= 1) {
    if ((int)threadIdx.x < off) red[threadIdx.x] += red[threadIdx.x + off];
    __syncthreads();
  }
  if (threadIdx.x == 0) den[row] = red[0];
}

// ---------------------------------------------------------------------------
// Kernel 3: num[b,d,e] = sum_s kT[b,d,s] * v[b,s,e].  grid = (D/32, D/128, B)
// ---------------------------------------------------------------------------
__global__ __launch_bounds__(256) void la_num_kernel(
    const __bf16* __restrict__ kT, const __bf16* __restrict__ v,
    float* __restrict__ num) {
  const int b = blockIdx.z;
  const int wave = threadIdx.x >> 5;
  const int lane = threadIdx.x & 31;
  const int half = lane >> 4;
  const int l16  = lane & 15;
  const int m0 = (blockIdx.y * 8 + wave) * 16;   // d
  const int n0 = blockIdx.x * 32;                // e
  const __bf16* A  = kT + (size_t)b * DD * SS;
  const __bf16* Bv = v  + (size_t)b * SS * DD + n0;

#if LA_HAVE_TDM
  __shared__ __bf16 tileV[KCH * 32];   // 8 KB bf16 v panel shared by 8 waves
#endif

  v8f c0 = {}; v8f c1 = {};
  for (int k0 = 0; k0 < SS; k0 += KCH) {
#if LA_HAVE_TDM
    __syncthreads();
    if (wave == 0) {
      tdm_load_2d((unsigned)(uintptr_t)tileV, Bv + (size_t)k0 * DD,
                  /*2B*/1, /*cols*/32, /*rows*/KCH, /*stride*/DD);
    }
    __syncthreads();
#endif
    if (k0 + KCH < SS)   // pull next A panel toward the WGP (global_prefetch_b8)
      __builtin_prefetch(A + (size_t)(m0 + l16) * SS + k0 + KCH, 0, 1);
#pragma unroll
    for (int kk = 0; kk < KCH; kk += 32) {
      v16bf a = load_a_bf16(A, SS, m0 + l16, k0 + kk, half);
#if LA_HAVE_TDM
      v16bf b0 = lds_b_frag_bf16(tileV, kk, l16, half);
      v16bf b1 = lds_b_frag_bf16(tileV, kk, l16 + 16, half);
#else
      v16bf b0 = load_b_gather_bf16(Bv - n0, DD, n0 + l16, k0 + kk, half);
      v16bf b1 = load_b_gather_bf16(Bv - n0, DD, n0 + 16 + l16, k0 + kk, half);
#endif
      c0 = wmma_bf16(a, b0, c0);
      c1 = wmma_bf16(a, b1, c1);
    }
  }

  float* C = num + (size_t)b * DD * DD;
#pragma unroll
  for (int r = 0; r < 8; ++r) {
    const int m = m0 + r + half * 8;
    C[(size_t)m * DD + n0 + l16]      = c0[r];
    C[(size_t)m * DD + n0 + 16 + l16] = c1[r];
  }
}

// ---------------------------------------------------------------------------
// Kernel 4: ctx = num / (den[e] + eps), cast to bf16.
// ---------------------------------------------------------------------------
__global__ __launch_bounds__(256) void la_ctx_kernel(
    const float* __restrict__ num, const float* __restrict__ den,
    __bf16* __restrict__ ctx) {
  const size_t i = (size_t)blockIdx.x * 256 + threadIdx.x;  // < B*D*D = 2^22
  const size_t e = i & (size_t)(DD - 1);
  const size_t b = i >> 20;                                 // D*D = 2^20
  ctx[i] = (__bf16)(num[i] / (den[b * DD + e] + 1e-6f));
}

// ---------------------------------------------------------------------------
// Kernel 5: out[b,d,s] = sum_e ctx[b,d,e] * q[b,s,e].  grid = (S/32, D/128, B)
// B[k=e][n=s] = q[s*D+e] is per-lane contiguous from global: keep direct path.
// ---------------------------------------------------------------------------
__global__ __launch_bounds__(256) void la_out_kernel(
    const __bf16* __restrict__ ctx, const __bf16* __restrict__ q,
    float* __restrict__ out) {
  const int b = blockIdx.z;
  const int wave = threadIdx.x >> 5;
  const int lane = threadIdx.x & 31;
  const int half = lane >> 4;
  const int l16  = lane & 15;
  const int m0 = (blockIdx.y * 8 + wave) * 16;   // d
  const int n0 = blockIdx.x * 32;                // s
  const __bf16* A = ctx + (size_t)b * DD * DD;
  const __bf16* Q = q   + (size_t)b * SS * DD;

  v8f c0 = {}; v8f c1 = {};
  for (int k0 = 0; k0 < DD; k0 += 32) {
    v16bf a  = load_a_bf16(A, DD, m0 + l16, k0, half);
    v16bf b0 = load_b_contig_bf16(Q, (size_t)(n0 + l16) * DD, k0, half);
    v16bf b1 = load_b_contig_bf16(Q, (size_t)(n0 + 16 + l16) * DD, k0, half);
    c0 = wmma_bf16(a, b0, c0);
    c1 = wmma_bf16(a, b1, c1);
  }

  float* O = out + (size_t)b * DD * SS;
#pragma unroll
  for (int r = 0; r < 8; ++r) {
    const int m = m0 + r + half * 8;
    O[(size_t)m * SS + n0 + l16]      = c0[r];
    O[(size_t)m * SS + n0 + 16 + l16] = c1[r];
  }
}

// ---------------------------------------------------------------------------
extern "C" void kernel_launch(void* const* d_in, const int* in_sizes, int n_in,
                              void* d_out, int out_size, void* d_ws, size_t ws_size,
                              hipStream_t stream) {
  const float* X  = (const float*)d_in[0];   // (B,S,E)
  const float* Wq = (const float*)d_in[1];   // (E,D)
  const float* Wk = (const float*)d_in[2];
  const float* Wv = (const float*)d_in[3];
  float* out = (float*)d_out;                // (B,D,S)

  char* ws = (char*)d_ws;
  __bf16* qO  = (__bf16*)(ws);
  __bf16* kT  = (__bf16*)(ws + ((size_t)32 << 20));
  __bf16* vO  = (__bf16*)(ws + ((size_t)64 << 20));
  float*  num = (float*) (ws + ((size_t)96 << 20));
  __bf16* ctx = (__bf16*)(ws + ((size_t)112 << 20));
  float*  den = (float*) (ws + ((size_t)120 << 20));

  la_qkv_kernel<<<dim3(DD / 32, (BB * SS) / 128, 3), 256, 0, stream>>>(
      X, Wq, Wk, Wv, qO, kT, vO);
  la_den_kernel<<<dim3(BB * DD), 256, 0, stream>>>(kT, den);
  la_num_kernel<<<dim3(DD / 32, DD / 128, BB), 256, 0, stream>>>(kT, vO, num);
  la_ctx_kernel<<<dim3((BB * DD * DD) / 256), 256, 0, stream>>>(num, den, ctx);
  la_out_kernel<<<dim3(SS / 32, DD / 128, BB), 256, 0, stream>>>(ctx, qO, out);
}
